// HOITransformerEncoder_56152402428154
// MI455X (gfx1250) — matrix operand order repaired and checked by reading, not compiled, not run on hardware
//
#include <hip/hip_runtime.h>
#include <math.h>

typedef __attribute__((ext_vector_type(16))) __bf16 bf16x16;
typedef __attribute__((ext_vector_type(8)))  __bf16 bf16x8;
typedef __attribute__((ext_vector_type(8)))  float  f32x8;

#define DIM   384
#define HEADS 6
#define HD    64
#define BATCH 8
#define NTOK  1024
#define MTOT  (BATCH*NTOK)   // 8192 rows (B*N)
#define FCD   1536
#define KNN   8
#define ATT_SCALE 0.125f     // HD^-0.5

// ---------------------------------------------------------------------------
// helpers
// ---------------------------------------------------------------------------
__device__ __forceinline__ bf16x16 ldfrag(const __bf16* lo, const __bf16* hi) {
  union { bf16x16 v; bf16x8 h[2]; } u;
  u.h[0] = *reinterpret_cast<const bf16x8*>(lo);
  u.h[1] = *reinterpret_cast<const bf16x8*>(hi);
  return u.v;
}

// ---------------------------------------------------------------------------
// Generic WMMA GEMM: out[M][Nn] = A[M][Kd] (bf16) x W[Nn][Kd]^T (bf16)
//   + bias (opt) + addsrc (opt, f32) ; act: 0=none 1=gelu(exact)
// Block: 128 threads (4 waves), tile 64(M) x 64(N), K-step 32.
// W tile staged through LDS. A/B fragments use contiguous k order, k = 16*half + s.
// ---------------------------------------------------------------------------
__global__ __launch_bounds__(128) void gemm_kernel(
    const __bf16* __restrict__ A, int lda,
    const __bf16* __restrict__ W, int Kd,
    const float*  __restrict__ bias,
    const float*  __restrict__ addsrc,
    float*        __restrict__ outF,
    __bf16*       __restrict__ outB,
    int Nn, int act)
{
  __shared__ __align__(16) __bf16 Bs[64 * 32];
  const int tid  = threadIdx.x;
  const int wave = tid >> 5;
  const int lane = tid & 31;
  const int half = lane >> 4;
  const int l15  = lane & 15;
  const int m0   = blockIdx.y * 64 + wave * 16;
  const int n0   = blockIdx.x * 64;

  f32x8 acc[4] = {};

  const int wrow = tid >> 1;          // 64 rows of W tile
  const int wcol = (tid & 1) * 16;    // two 16-elem halves of the 32-k slab
  const __bf16* wptr = W + (size_t)(n0 + wrow) * Kd + wcol;
  const __bf16* arow = A + (size_t)(m0 + l15) * lda + half * 16;

  for (int k0 = 0; k0 < Kd; k0 += 32) {
    __syncthreads();
    *reinterpret_cast<bf16x8*>(&Bs[wrow * 32 + wcol])     = *reinterpret_cast<const bf16x8*>(wptr + k0);
    *reinterpret_cast<bf16x8*>(&Bs[wrow * 32 + wcol + 8]) = *reinterpret_cast<const bf16x8*>(wptr + k0 + 8);
    if (k0 + 32 < Kd) __builtin_prefetch(wptr + k0 + 32, 0, 0);
    __syncthreads();

    bf16x16 af = ldfrag(arow + k0, arow + k0 + 8);
#pragma unroll
    for (int t = 0; t < 4; ++t) {
      const __bf16* bsrc = &Bs[(t * 16 + l15) * 32 + half * 16];
      bf16x16 bv = ldfrag(bsrc, bsrc + 8);
      acc[t] = __builtin_amdgcn_wmma_f32_16x16x32_bf16(false, af, false, bv,
                                                       (short)0, acc[t], false, false);
    }
  }

#pragma unroll
  for (int t = 0; t < 4; ++t) {
#pragma unroll
    for (int j = 0; j < 8; ++j) {
      const int row = m0 + j + half * 8;
      const int col = n0 + t * 16 + l15;
      float x = acc[t][j];
      if (bias)   x += bias[col];
      if (addsrc) x += addsrc[(size_t)row * Nn + col];
      if (act == 1) x = 0.5f * x * (1.0f + erff(x * 0.70710678118f));
      const size_t o = (size_t)row * Nn + col;
      if (outF) outF[o] = x;
      if (outB) outB[o] = (__bf16)x;
    }
  }
}

// ---------------------------------------------------------------------------
// Flash attention: one wave = 16 query rows, head-dim 64, online softmax.
// Q,K row-major bf16 with row stride ldq/ldk and per-head column base.
// Vt: [B*H][64][nk] bf16 (transposed V). Out bf16 [B*nq][ldo], cols h*64..
// ---------------------------------------------------------------------------
__global__ __launch_bounds__(128) void attn_kernel(
    const __bf16* __restrict__ Qb, int ldq, int qbase,
    const __bf16* __restrict__ Kb, int ldk, int kbase,
    const __bf16* __restrict__ Vt,
    __bf16* __restrict__ Ob, int ldo,
    int nq, int nk)
{
  __shared__ __align__(16) __bf16 Pl[4][16 * 32];
  const int tid  = threadIdx.x;
  const int wave = tid >> 5;
  const int lane = tid & 31;
  const int half = lane >> 4;
  const int l15  = lane & 15;
  const int b = blockIdx.z, h = blockIdx.y;
  const int q0 = blockIdx.x * 64 + wave * 16;

  const __bf16* qrow = Qb + (size_t)(b * nq + q0 + l15) * ldq + qbase + h * HD + half * 16;
  bf16x16 qa0 = ldfrag(qrow,      qrow + 8);
  bf16x16 qa1 = ldfrag(qrow + 32, qrow + 40);

  f32x8 oc[4] = {};
  float mrun[8], lrun[8];
#pragma unroll
  for (int j = 0; j < 8; ++j) { mrun[j] = -__builtin_inff(); lrun[j] = 0.f; }

  const __bf16* kcol = Kb + (size_t)(b * nk) * ldk + kbase + h * HD;
  const __bf16* vrow = Vt + (size_t)(b * HEADS + h) * HD * nk;

  for (int kb0 = 0; kb0 < nk; kb0 += 32) {
    __syncthreads();
    f32x8 s0 = {}, s1 = {};
    {
      const __bf16* krA = kcol + (size_t)(kb0 + l15) * ldk + half * 16;
      const __bf16* krB = kcol + (size_t)(kb0 + 16 + l15) * ldk + half * 16;
      bf16x16 f0a = ldfrag(krA,      krA + 8);
      bf16x16 f1a = ldfrag(krA + 32, krA + 40);
      bf16x16 f0b = ldfrag(krB,      krB + 8);
      bf16x16 f1b = ldfrag(krB + 32, krB + 40);
      s0 = __builtin_amdgcn_wmma_f32_16x16x32_bf16(false, qa0, false, f0a, (short)0, s0, false, false);
      s0 = __builtin_amdgcn_wmma_f32_16x16x32_bf16(false, qa1, false, f1a, (short)0, s0, false, false);
      s1 = __builtin_amdgcn_wmma_f32_16x16x32_bf16(false, qa0, false, f0b, (short)0, s1, false, false);
      s1 = __builtin_amdgcn_wmma_f32_16x16x32_bf16(false, qa1, false, f1b, (short)0, s1, false, false);
    }

    float p0[8], p1[8], sc[8];
#pragma unroll
    for (int j = 0; j < 8; ++j) {
      float a0 = s0[j] * ATT_SCALE, a1 = s1[j] * ATT_SCALE;
      float cm = fmaxf(a0, a1);
#pragma unroll
      for (int m = 1; m < 16; m <<= 1) cm = fmaxf(cm, __shfl_xor(cm, m));
      float mn  = fmaxf(mrun[j], cm);
      float scl = __expf(mrun[j] - mn);
      p0[j] = __expf(a0 - mn);
      p1[j] = __expf(a1 - mn);
      float rs = p0[j] + p1[j];
#pragma unroll
      for (int m = 1; m < 16; m <<= 1) rs += __shfl_xor(rs, m);
      lrun[j] = lrun[j] * scl + rs;
      mrun[j] = mn;
      sc[j] = scl;
    }
#pragma unroll
    for (int t = 0; t < 4; ++t)
#pragma unroll
      for (int j = 0; j < 8; ++j) oc[t][j] *= sc[j];

    __bf16* pl = Pl[wave];
#pragma unroll
    for (int j = 0; j < 8; ++j) {
      pl[(j + half * 8) * 32 + l15]      = (__bf16)p0[j];
      pl[(j + half * 8) * 32 + l15 + 16] = (__bf16)p1[j];
    }
    __syncthreads();
    const __bf16* pr = &pl[l15 * 32 + half * 16];
    bf16x16 pa = ldfrag(pr, pr + 8);
#pragma unroll
    for (int t = 0; t < 4; ++t) {
      const __bf16* vv = vrow + (size_t)(t * 16 + l15) * nk + kb0 + half * 16;
      bf16x16 vb = ldfrag(vv, vv + 8);
      oc[t] = __builtin_amdgcn_wmma_f32_16x16x32_bf16(false, pa, false, vb, (short)0, oc[t], false, false);
    }
  }

#pragma unroll
  for (int t = 0; t < 4; ++t)
#pragma unroll
    for (int j = 0; j < 8; ++j) {
      const int row = q0 + j + half * 8;
      const int col = h * HD + t * 16 + l15;
      Ob[(size_t)(b * nq + row) * ldo + col] = (__bf16)(oc[t][j] / lrun[j]);
    }
}

// ---------------------------------------------------------------------------
// LayerNorm: one wave per 384-elem row, 8 rows/block, bf16 output.
// ---------------------------------------------------------------------------
__global__ __launch_bounds__(256) void ln_kernel(
    const float* __restrict__ x, const float* __restrict__ w,
    const float* __restrict__ b, __bf16* __restrict__ out, int rows)
{
  const int wave = threadIdx.x >> 5, lane = threadIdx.x & 31;
  const size_t row = (size_t)blockIdx.x * 8 + wave;
  if (row >= (size_t)rows) return;
  const float* xr = x + row * DIM;
  float v[12], s = 0.f, s2 = 0.f;
#pragma unroll
  for (int i = 0; i < 12; ++i) { v[i] = xr[lane + i * 32]; s += v[i]; s2 += v[i] * v[i]; }
#pragma unroll
  for (int m = 1; m < 32; m <<= 1) { s += __shfl_xor(s, m); s2 += __shfl_xor(s2, m); }
  float mean = s * (1.f / DIM);
  float var  = s2 * (1.f / DIM) - mean * mean;
  float r    = rsqrtf(var + 1e-5f);
  __bf16* orow = out + row * DIM;
#pragma unroll
  for (int i = 0; i < 12; ++i) {
    int c = lane + i * 32;
    orow[c] = (__bf16)((v[i] - mean) * r * w[c] + b[c]);
  }
}

// ---------------------------------------------------------------------------
// Brute-force KNN top-8 (3D points). cq/ck: [B][3][n]; idx out: [B][8][nq]
// ---------------------------------------------------------------------------
__global__ __launch_bounds__(256) void knn_idx_kernel(
    const float* __restrict__ cq, const float* __restrict__ ck,
    int* __restrict__ idx, int nq, int nk)
{
  int gid = blockIdx.x * 256 + threadIdx.x;
  if (gid >= BATCH * nq) return;
  int b = gid / nq, n = gid % nq;
  const float* qp = cq + (size_t)b * 3 * nq;
  float qx = qp[n], qy = qp[nq + n], qz = qp[2 * nq + n];
  const float* kp = ck + (size_t)b * 3 * nk;
  float bd[KNN]; int bi[KNN];
#pragma unroll
  for (int k = 0; k < KNN; ++k) { bd[k] = __builtin_inff(); bi[k] = 0; }
  for (int m = 0; m < nk; ++m) {
    float dx = qx - kp[m], dy = qy - kp[nk + m], dz = qz - kp[2 * nk + m];
    float d = dx * dx + dy * dy + dz * dz;
    if (d < bd[KNN - 1]) {
#pragma unroll
      for (int k = KNN - 1; k >= 1; --k) {
        if (d < bd[k]) {
          if (d < bd[k - 1]) { bd[k] = bd[k - 1]; bi[k] = bi[k - 1]; }
          else               { bd[k] = d;         bi[k] = m;         }
        }
      }
      if (d < bd[0]) { bd[0] = d; bi[0] = m; }
    }
  }
#pragma unroll
  for (int k = 0; k < KNN; ++k) idx[((size_t)b * KNN + k) * nq + n] = bi[k];
}

// out[n][c] = max_k leakyrelu( h[idx[k][n]][c] + g[n][c] )
__global__ __launch_bounds__(256) void knn_max_kernel(
    const __bf16* __restrict__ hsrc, const float* __restrict__ g,
    const int* __restrict__ idx, __bf16* __restrict__ out, int ntok)
{
  size_t gid = (size_t)blockIdx.x * 256 + threadIdx.x;
  if (gid >= (size_t)BATCH * ntok * DIM) return;
  int c = (int)(gid % DIM);
  size_t rn = gid / DIM;
  int n = (int)(rn % ntok);
  int b = (int)(rn / ntok);
  float gv = g[rn * DIM + c];
  float best = -__builtin_inff();
#pragma unroll
  for (int k = 0; k < KNN; ++k) {
    int m = idx[((size_t)b * KNN + k) * ntok + n];
    float x = (float)hsrc[((size_t)b * ntok + m) * DIM + c] + gv;
    x = (x > 0.f) ? x : 0.2f * x;
    best = fmaxf(best, x);
  }
  out[rn * DIM + c] = (__bf16)best;
}

// transpose-pack V heads: vt[b][h][d][m] = src[b*nk+m][coff + h*64 + d]
__global__ __launch_bounds__(256) void pack_vt_kernel(
    const __bf16* __restrict__ src, int lds_, int coff,
    __bf16* __restrict__ vt, int nk)
{
  size_t gid = (size_t)blockIdx.x * 256 + threadIdx.x;
  if (gid >= (size_t)BATCH * HEADS * HD * nk) return;
  int m = (int)(gid % nk);
  size_t r = gid / nk;
  int d = (int)(r % HD);
  size_t bh = r / HD;
  int h = (int)(bh % HEADS);
  int b = (int)(bh / HEADS);
  vt[gid] = src[((size_t)b * nk + m) * lds_ + coff + h * HD + d];
}

// weight convert f32 -> bf16, optional column window and W2-W1 diff
__global__ __launch_bounds__(256) void wconv_kernel(
    const float* __restrict__ w, __bf16* __restrict__ out,
    int nn, int kd, int ldin, int offc, int diff)
{
  size_t gid = (size_t)blockIdx.x * 256 + threadIdx.x;
  if (gid >= (size_t)nn * kd) return;
  int k = (int)(gid % kd);
  int n = (int)(gid / kd);
  float x = w[(size_t)n * ldin + offc + k];
  if (diff) x -= w[(size_t)n * ldin + k];
  out[gid] = (__bf16)x;
}

__global__ __launch_bounds__(256) void copy_kernel(const float* __restrict__ a,
                                                   float* __restrict__ o, size_t n)
{
  size_t i = (size_t)blockIdx.x * 256 + threadIdx.x;
  if (i < n) o[i] = a[i];
}

__global__ __launch_bounds__(256) void add2_kernel(float* __restrict__ q,
                                                   const float* __restrict__ a,
                                                   const float* __restrict__ b,
                                                   size_t n)
{
  size_t i = (size_t)blockIdx.x * 256 + threadIdx.x;
  if (i < n) q[i] += a[i] + b[i];
}

// ---------------------------------------------------------------------------
static inline void* wsalloc(char* base, size_t& off, size_t bytes) {
  void* p = base + off;
  off += (bytes + 255) & ~(size_t)255;
  return p;
}

extern "C" void kernel_launch(void* const* d_in, const int* in_sizes, int n_in,
                              void* d_out, int out_size, void* d_ws, size_t ws_size,
                              hipStream_t stream)
{
  (void)in_sizes; (void)n_in; (void)out_size; (void)ws_size;
  const float* q_in = (const float*)d_in[0];
  const float* v_in = (const float*)d_in[1];
  const float* cq   = (const float*)d_in[2];
  const float* cv   = (const float*)d_in[3];
  // params flattened in _make_params insertion order: 28 tensors per block
#define PB(i, j) ((const float*)d_in[4 + (i) * 28 + (j)])

  char* base = (char*)d_ws;
  size_t off = 0;
  const size_t ND  = (size_t)MTOT * DIM;

  float*  qf   = (float*)wsalloc(base, off, ND * 4);
  float*  q1f  = (float*)wsalloc(base, off, ND * 4);
  float*  q2f  = (float*)wsalloc(base, off, ND * 4);
  float*  gS   = (float*)wsalloc(base, off, ND * 4);
  float*  gC   = (float*)wsalloc(base, off, ND * 4);
  __bf16* nq1b = (__bf16*)wsalloc(base, off, ND * 2);
  __bf16* nqb  = (__bf16*)wsalloc(base, off, ND * 2);
  __bf16* nvb  = (__bf16*)wsalloc(base, off, ND * 2);
  __bf16* n2b  = (__bf16*)wsalloc(base, off, ND * 2);
  __bf16* attnb= (__bf16*)wsalloc(base, off, ND * 2);
  __bf16* cab  = (__bf16*)wsalloc(base, off, ND * 2);
  __bf16* q1b  = (__bf16*)wsalloc(base, off, ND * 2);
  __bf16* q2b  = (__bf16*)wsalloc(base, off, ND * 2);
  __bf16* cqb  = (__bf16*)wsalloc(base, off, ND * 2);
  __bf16* ckb  = (__bf16*)wsalloc(base, off, ND * 2);
  __bf16* cvb  = (__bf16*)wsalloc(base, off, ND * 2);
  __bf16* hSb  = (__bf16*)wsalloc(base, off, ND * 2);
  __bf16* hCb  = (__bf16*)wsalloc(base, off, ND * 2);
  __bf16* fSb  = (__bf16*)wsalloc(base, off, ND * 2);
  __bf16* fCb  = (__bf16*)wsalloc(base, off, ND * 2);
  __bf16* qkvb = (__bf16*)wsalloc(base, off, (size_t)MTOT * 1152 * 2);
  __bf16* vtb  = (__bf16*)wsalloc(base, off, (size_t)BATCH * HEADS * HD * NTOK * 2);
  __bf16* fc1b = (__bf16*)wsalloc(base, off, (size_t)MTOT * FCD * 2);
  int* idxS = (int*)wsalloc(base, off, (size_t)BATCH * KNN * NTOK * 4);
  int* idxC = (int*)wsalloc(base, off, (size_t)BATCH * KNN * NTOK * 4);

  __bf16 *Wqkv[4], *Wproj[4], *Wcq[4], *Wck[4], *Wcv[4], *Wcproj[4], *Wfc1[4], *Wfc2[4];
  for (int i = 0; i < 4; ++i) {
    Wqkv[i]   = (__bf16*)wsalloc(base, off, (size_t)1152 * 384 * 2);
    Wproj[i]  = (__bf16*)wsalloc(base, off, (size_t)384 * 384 * 2);
    Wcq[i]    = (__bf16*)wsalloc(base, off, (size_t)384 * 384 * 2);
    Wck[i]    = (__bf16*)wsalloc(base, off, (size_t)384 * 384 * 2);
    Wcv[i]    = (__bf16*)wsalloc(base, off, (size_t)384 * 384 * 2);
    Wcproj[i] = (__bf16*)wsalloc(base, off, (size_t)384 * 384 * 2);
    Wfc1[i]   = (__bf16*)wsalloc(base, off, (size_t)1536 * 384 * 2);
    Wfc2[i]   = (__bf16*)wsalloc(base, off, (size_t)384 * 1536 * 2);
  }
  __bf16* Wkn1 = (__bf16*)wsalloc(base, off, (size_t)384 * 384 * 2);
  __bf16* Wknd = (__bf16*)wsalloc(base, off, (size_t)384 * 384 * 2);
  __bf16* Wm1  = (__bf16*)wsalloc(base, off, (size_t)384 * 384 * 2);
  __bf16* Wm2  = (__bf16*)wsalloc(base, off, (size_t)384 * 384 * 2);
  __bf16* Wck1 = (__bf16*)wsalloc(base, off, (size_t)384 * 384 * 2);
  __bf16* Wckd = (__bf16*)wsalloc(base, off, (size_t)384 * 384 * 2);
  __bf16* Wcm1 = (__bf16*)wsalloc(base, off, (size_t)384 * 384 * 2);
  __bf16* Wcm2 = (__bf16*)wsalloc(base, off, (size_t)384 * 384 * 2);

  auto wc = [&](const float* w, __bf16* o, int nn, int kd, int ldin, int offc, int diff) {
    size_t n = (size_t)nn * kd;
    wconv_kernel<<<dim3((unsigned)((n + 255) / 256)), dim3(256), 0, stream>>>(w, o, nn, kd, ldin, offc, diff);
  };
  for (int i = 0; i < 4; ++i) {
    wc(PB(i, 8),  Wqkv[i],  1152, 384, 384, 0, 0);
    wc(PB(i, 9),  Wproj[i],  384, 384, 384, 0, 0);
    wc(PB(i, 11), Wcq[i],    384, 384, 384, 0, 0);
    wc(PB(i, 12), Wck[i],    384, 384, 384, 0, 0);
    wc(PB(i, 13), Wcv[i],    384, 384, 384, 0, 0);
    wc(PB(i, 14), Wcproj[i], 384, 384, 384, 0, 0);
    wc(PB(i, 24), Wfc1[i],  1536, 384, 384, 0, 0);
    wc(PB(i, 26), Wfc2[i],   384, 1536, 1536, 0, 0);
  }
  wc(PB(0, 16), Wkn1, 384, 384, 768, 0,   0);
  wc(PB(0, 16), Wknd, 384, 384, 768, 384, 1);
  wc(PB(0, 18), Wm1,  384, 384, 768, 0,   0);
  wc(PB(0, 18), Wm2,  384, 384, 768, 384, 0);
  wc(PB(0, 20), Wck1, 384, 384, 768, 0,   0);
  wc(PB(0, 20), Wckd, 384, 384, 768, 384, 1);
  wc(PB(0, 22), Wcm1, 384, 384, 768, 0,   0);
  wc(PB(0, 22), Wcm2, 384, 384, 768, 384, 0);

  copy_kernel<<<dim3((unsigned)((ND + 255) / 256)), dim3(256), 0, stream>>>(q_in, qf, ND);

  knn_idx_kernel<<<dim3((BATCH * NTOK + 255) / 256), dim3(256), 0, stream>>>(cq, cq, idxS, NTOK, NTOK);
  knn_idx_kernel<<<dim3((BATCH * NTOK + 255) / 256), dim3(256), 0, stream>>>(cq, cv, idxC, NTOK, NTOK);

  auto gemm = [&](const __bf16* A, int lda, const __bf16* W, int Kd, int Nn,
                  const float* bias, const float* add, float* oF, __bf16* oB, int act) {
    dim3 g(Nn / 64, MTOT / 64);
    gemm_kernel<<<g, dim3(128), 0, stream>>>(A, lda, W, Kd, bias, add, oF, oB, Nn, act);
  };
  auto attn = [&](const __bf16* Qb, int ldq, int qb0, const __bf16* Kb, int ldk, int kb0,
                  const __bf16* Vt, __bf16* Ob) {
    dim3 g(NTOK / 64, HEADS, BATCH);
    attn_kernel<<<g, dim3(128), 0, stream>>>(Qb, ldq, qb0, Kb, ldk, kb0, Vt, Ob, DIM, NTOK, NTOK);
  };
  auto ln = [&](const float* x, const float* w, const float* b, __bf16* o) {
    ln_kernel<<<dim3(MTOT / 8), dim3(256), 0, stream>>>(x, w, b, o, MTOT);
  };
  const unsigned g256 = (unsigned)((ND + 255) / 256);
  const unsigned gvt  = (unsigned)(((size_t)BATCH * HEADS * HD * NTOK + 255) / 256);

  for (int i = 0; i < 4; ++i) {
    // --- self attention branch ---
    ln(qf, PB(i, 0), PB(i, 1), nq1b);
    gemm(nq1b, DIM, Wqkv[i], DIM, 1152, nullptr, nullptr, nullptr, qkvb, 0);
    pack_vt_kernel<<<dim3(gvt), dim3(256), 0, stream>>>(qkvb, 1152, 768, vtb, NTOK);
    attn(qkvb, 1152, 0, qkvb, 1152, 384, vtb, attnb);
    gemm(attnb, DIM, Wproj[i], DIM, DIM, PB(i, 10), nullptr, q1f, (i == 0 ? q1b : nullptr), 0);

    // --- cross attention branch ---
    ln(qf,   PB(i, 4), PB(i, 5), nqb);
    ln(v_in, PB(i, 6), PB(i, 7), nvb);
    gemm(nqb, DIM, Wcq[i], DIM, DIM, nullptr, nullptr, nullptr, cqb, 0);
    gemm(nvb, DIM, Wck[i], DIM, DIM, nullptr, nullptr, nullptr, ckb, 0);
    gemm(nvb, DIM, Wcv[i], DIM, DIM, nullptr, nullptr, nullptr, cvb, 0);
    pack_vt_kernel<<<dim3(gvt), dim3(256), 0, stream>>>(cvb, DIM, 0, vtb, NTOK);
    attn(cqb, DIM, 0, ckb, DIM, 0, vtb, cab);
    gemm(cab, DIM, Wcproj[i], DIM, DIM, PB(i, 15), nullptr, q2f, (i == 0 ? q2b : nullptr), 0);

    if (i == 0) {
      // self KNN: f = max_k lrelu( nq[idx]@W1^T + nq@(W2-W1)^T + b )
      gemm(nqb, DIM, Wkn1, DIM, DIM, nullptr, nullptr, nullptr, hSb, 0);
      gemm(nqb, DIM, Wknd, DIM, DIM, PB(0, 17), nullptr, gS, nullptr, 0);
      knn_max_kernel<<<dim3(g256), dim3(256), 0, stream>>>(hSb, gS, idxS, fSb, NTOK);
      gemm(q1b, DIM, Wm1, DIM, DIM, PB(0, 19), nullptr, q1f, nullptr, 0);
      gemm(fSb, DIM, Wm2, DIM, DIM, nullptr, q1f, q1f, nullptr, 0);
      // cross KNN: nb from nv, ctr = nq
      gemm(nvb, DIM, Wck1, DIM, DIM, nullptr, nullptr, nullptr, hCb, 0);
      gemm(nqb, DIM, Wckd, DIM, DIM, PB(0, 21), nullptr, gC, nullptr, 0);
      knn_max_kernel<<<dim3(g256), dim3(256), 0, stream>>>(hCb, gC, idxC, fCb, NTOK);
      gemm(q2b, DIM, Wcm1, DIM, DIM, PB(0, 23), nullptr, q2f, nullptr, 0);
      gemm(fCb, DIM, Wcm2, DIM, DIM, nullptr, q2f, q2f, nullptr, 0);
    }

    add2_kernel<<<dim3(g256), dim3(256), 0, stream>>>(qf, q1f, q2f, ND);

    // --- MLP ---
    ln(qf, PB(i, 2), PB(i, 3), n2b);
    gemm(n2b, DIM, Wfc1[i], DIM, FCD, PB(i, 25), nullptr, nullptr, fc1b, 1);
    gemm(fc1b, FCD, Wfc2[i], FCD, DIM, PB(i, 27), qf, qf, nullptr, 0);
  }

  copy_kernel<<<dim3(g256), dim3(256), 0, stream>>>(qf, (float*)d_out, ND);
#undef PB
}
